// AttentionModel_39848706572895
// MI455X (gfx1250) — matrix operand order repaired and checked by reading, not compiled, not run on hardware
//
#include <hip/hip_runtime.h>

typedef __attribute__((ext_vector_type(16))) _Float16 v16h;
typedef __attribute__((ext_vector_type(8)))  _Float16 h8;
typedef __attribute__((ext_vector_type(8)))  float    v8f;

#define SQ 2048
#define NH 12
#define HD 64
#define NBH 24
#define ROWSTRIDE (NH * HD)   // 768 floats between s rows in [B,S,H,D]

__device__ __forceinline__ unsigned hash32(unsigned x) {
  x ^= x >> 16; x *= 0x7feb352dU;
  x ^= x >> 15; x *= 0x846ca68bU;
  x ^= x >> 16;
  return x;
}

// 3-op generator: 4 dropout bytes per call
__device__ __forceinline__ unsigned rng4(unsigned x) {
  x *= 2654435761u;
  x ^= x >> 16;
  x *= 2246822519u;
  return x;
}

__device__ __forceinline__ v16h cat8(h8 a, h8 b) {
  return __builtin_shufflevector(a, b, 0, 1, 2, 3, 4, 5, 6, 7,
                                 8, 9, 10, 11, 12, 13, 14, 15);
}

__device__ __forceinline__ h8 cvt8(float4 a, float4 b) {
  h8 r;
  r[0] = (_Float16)a.x; r[1] = (_Float16)a.y; r[2] = (_Float16)a.z; r[3] = (_Float16)a.w;
  r[4] = (_Float16)b.x; r[5] = (_Float16)b.y; r[6] = (_Float16)b.z; r[7] = (_Float16)b.w;
  return r;
}

// B-operand (16-bit B 32x16 layout) from global fp32: lane col n = lane&15,
// k = 8*hi + e (e<8), 16 + 8*hi + (e-8) (e>=8) -> two 8-float runs.
__device__ __forceinline__ v16h load_run16(const float* __restrict__ p, float scl) {
  float4 a0 = *(const float4*)(p);
  float4 a1 = *(const float4*)(p + 4);
  float4 a2 = *(const float4*)(p + 16);
  float4 a3 = *(const float4*)(p + 20);
  v16h r;
  r[0]  = (_Float16)(a0.x * scl); r[1]  = (_Float16)(a0.y * scl);
  r[2]  = (_Float16)(a0.z * scl); r[3]  = (_Float16)(a0.w * scl);
  r[4]  = (_Float16)(a1.x * scl); r[5]  = (_Float16)(a1.y * scl);
  r[6]  = (_Float16)(a1.z * scl); r[7]  = (_Float16)(a1.w * scl);
  r[8]  = (_Float16)(a2.x * scl); r[9]  = (_Float16)(a2.y * scl);
  r[10] = (_Float16)(a2.z * scl); r[11] = (_Float16)(a2.w * scl);
  r[12] = (_Float16)(a3.x * scl); r[13] = (_Float16)(a3.y * scl);
  r[14] = (_Float16)(a3.z * scl); r[15] = (_Float16)(a3.w * scl);
  return r;
}

// ===================== pre-pass: fp32 [B,S,H,D] -> f16 workspace ============
// Kh[bh][s][d] (row-major f16) and Vt[bh][d][s] (transposed f16).
// One 64s x 64d tile per block; transpose through LDS.
__global__ void __launch_bounds__(256)
preconv(const float* __restrict__ K, const float* __restrict__ V,
        _Float16* __restrict__ Kh, _Float16* __restrict__ Vt) {
  const int bh = blockIdx.y;
  const int b = bh / NH, h = bh % NH;
  const int s0 = blockIdx.x << 6;
  const int tid = threadIdx.x;
  const int srow = tid >> 2;             // 0..63
  const int dseg = (tid & 3) << 4;       // 0,16,32,48

  __shared__ __align__(16) _Float16 T[64][72];

  const float* kin = K + ((size_t)b * SQ + s0 + srow) * ROWSTRIDE + h * HD + dseg;
  const float* vin = V + ((size_t)b * SQ + s0 + srow) * ROWSTRIDE + h * HD + dseg;

  float4 a0 = *(const float4*)(kin);
  float4 a1 = *(const float4*)(kin + 4);
  float4 a2 = *(const float4*)(kin + 8);
  float4 a3 = *(const float4*)(kin + 12);
  _Float16* kout = Kh + ((size_t)bh * SQ + s0 + srow) * HD + dseg;
  *(h8*)(kout)     = cvt8(a0, a1);
  *(h8*)(kout + 8) = cvt8(a2, a3);

  float4 b0 = *(const float4*)(vin);
  float4 b1 = *(const float4*)(vin + 4);
  float4 b2 = *(const float4*)(vin + 8);
  float4 b3 = *(const float4*)(vin + 12);
  *(h8*)&T[srow][dseg]     = cvt8(b0, b1);
  *(h8*)&T[srow][dseg + 8] = cvt8(b2, b3);

  __syncthreads();

  const int drow = tid >> 2;             // 0..63
  const int sseg = (tid & 3) << 4;       // 0,16,32,48
  h8 o0, o1;
#pragma unroll
  for (int j = 0; j < 8; ++j) { o0[j] = T[sseg + j][drow]; o1[j] = T[sseg + 8 + j][drow]; }
  _Float16* vout = Vt + ((size_t)bh * HD + drow) * SQ + s0 + sseg;
  *(h8*)(vout)     = o0;
  *(h8*)(vout + 8) = o1;
}

// ===================== main kernel: f16 operands straight from memory =======
// One wave = one 16-row Q tile; 64 keys per iteration; 16 WMMAs per iteration.
__global__ void __launch_bounds__(256)
fattn_f16(const float* __restrict__ Q, const _Float16* __restrict__ Kh,
          const _Float16* __restrict__ Vt, float* __restrict__ O) {
  const int lane  = threadIdx.x & 31;
  const int wid   = threadIdx.x >> 5;
  const int lan16 = lane & 15;
  const int hi    = lane >> 4;

  const int tile = blockIdx.x * 8 + wid;   // qt fastest -> blocks share a head's KV in L2
  const int qt = tile & 127;
  const int bh = tile >> 7;
  const int b = bh / NH, h = bh % NH;

  const _Float16* kbase = Kh + (size_t)bh * SQ * HD;
  const _Float16* vbase = Vt + (size_t)bh * HD * SQ;
  const float*    qbase = Q + (size_t)b * SQ * ROWSTRIDE + h * HD;
  float*          obase = O + ((size_t)b * NH + h) * SQ * HD;   // [B,H,S,D]

  const int q0 = qt * 16;
  const float qscale = 0.125f * 1.44269504f;   // 1/sqrt(D) * log2(e): exp2-domain softmax
  const float* qrow = qbase + (size_t)(q0 + lan16) * ROWSTRIDE;
  const v16h bQ0 = load_run16(qrow + 0  + 8 * hi, qscale);
  const v16h bQ1 = load_run16(qrow + 32 + 8 * hi, qscale);

  v8f acc[4] = {{0}, {0}, {0}, {0}};
  float m_run = -1.0e30f;
  float l_run = 0.0f;
  const float inv_keep = 1.0f / 0.9f;
  const unsigned seed_base = ((((unsigned)bh << 11) | (unsigned)(q0 + lan16)) << 11);

#pragma unroll 1
  for (int t0 = 0; t0 < SQ; t0 += 64) {
    // ---- scores: four 16x16 S^T sub-tiles (A = K rows, B = Q) ----
    v8f s[4];
#pragma unroll
    for (int st = 0; st < 4; ++st) {
      const _Float16* kr = kbase + (size_t)(t0 + st * 16 + lan16) * HD;
      const v16h a0 = cat8(*(const h8*)(kr + 8 * hi),      *(const h8*)(kr + 16 + 8 * hi));
      const v16h a1 = cat8(*(const h8*)(kr + 32 + 8 * hi), *(const h8*)(kr + 48 + 8 * hi));
      v8f t = {};
      t = __builtin_amdgcn_wmma_f32_16x16x32_f16(false, a0, false, bQ0, (short)0, t, false, false);
      t = __builtin_amdgcn_wmma_f32_16x16x32_f16(false, a1, false, bQ1, (short)0, t, false, false);
      s[st] = t;
    }
    if (t0 + 64 < SQ) {
      __builtin_prefetch(kbase + (size_t)(t0 + 64 + lan16) * HD, 0, 1);
      __builtin_prefetch(vbase + (size_t)lan16 * SQ + t0 + 64, 0, 1);
    }

    // ---- online softmax (exp2 domain); lane owns q-row lane&15;
    //      s[st][r] is t = t0 + st*16 + 8*hi + r ----
    float mt = s[0][0];
#pragma unroll
    for (int st = 0; st < 4; ++st)
#pragma unroll
      for (int r = 0; r < 8; ++r) mt = fmaxf(mt, s[st][r]);
    mt = fmaxf(mt, __shfl_xor(mt, 16, 32));
    const float m_new = fmaxf(m_run, mt);

    // rescale only when some row's max actually moved (wave-uniform branch)
    float corr = 1.0f;
    if (__ballot(m_new > m_run)) {
      corr = __builtin_amdgcn_exp2f(m_run - m_new);
#pragma unroll
      for (int r = 0; r < 8; ++r) {
        const float c = __shfl(corr, r + 8 * hi, 32);
        acc[0][r] *= c; acc[1][r] *= c; acc[2][r] *= c; acc[3][r] *= c;
      }
    }

    // ---- probabilities + dropout; results land directly in A-operand layout ----
    float lt = 0.0f;
    v16h aP0, aP1;   // t rel 0..31 and 32..63
#pragma unroll
    for (int st = 0; st < 4; ++st) {
      const unsigned ha = rng4(seed_base + (unsigned)(t0 + st * 16 + 8 * hi));
      const unsigned hb = rng4(seed_base + (unsigned)(t0 + st * 16 + 8 * hi + 4));
#pragma unroll
      for (int r = 0; r < 8; ++r) {
        const float pr = __builtin_amdgcn_exp2f(s[st][r] - m_new);
        lt += pr;
        const unsigned hh = (r < 4) ? ha : hb;
        const float mk = (((hh >> (8 * (r & 3))) & 255u) < 230u) ? inv_keep : 0.0f;
        const _Float16 pd = (_Float16)(pr * mk);
        if (st == 0) aP0[r] = pd;
        else if (st == 1) aP0[8 + r] = pd;
        else if (st == 2) aP1[r] = pd;
        else aP1[8 + r] = pd;
      }
    }
    lt += __shfl_xor(lt, 16, 32);
    l_run = l_run * corr + lt;
    m_run = m_new;

    // ---- P(16x64) x V(64x16) per 16-wide d chunk; B from transposed Vt ----
#pragma unroll
    for (int dn = 0; dn < 4; ++dn) {
      const _Float16* vr = vbase + (size_t)(dn * 16 + lan16) * SQ + t0 + 8 * hi;
      const v16h b0 = cat8(*(const h8*)(vr),      *(const h8*)(vr + 16));
      const v16h b1 = cat8(*(const h8*)(vr + 32), *(const h8*)(vr + 48));
      acc[dn] = __builtin_amdgcn_wmma_f32_16x16x32_f16(false, aP0, false, b0, (short)0, acc[dn], false, false);
      acc[dn] = __builtin_amdgcn_wmma_f32_16x16x32_f16(false, aP1, false, b1, (short)0, acc[dn], false, false);
    }
  }

  // ---- epilogue ----
#pragma unroll
  for (int r = 0; r < 8; ++r) {
    const float linv = 1.0f / __shfl(l_run, r + 8 * hi, 32);
    float* orow = obase + (size_t)(q0 + r + 8 * hi) * HD + lan16;
    orow[0]  = acc[0][r] * linv;
    orow[16] = acc[1][r] * linv;
    orow[32] = acc[2][r] * linv;
    orow[48] = acc[3][r] * linv;
  }
}

// ===================== fallback (self-contained, LDS-staged) ================
#define TS 32
#define NIT (SQ / TS)
#define KPAD 72
#define VPAD 40

__device__ __forceinline__ v16h ldsA(const _Float16* row, int cbase, int hi) {
  h8 lo = *(const h8*)(row + cbase + 8 * hi);
  h8 hh = *(const h8*)(row + cbase + 16 + 8 * hi);
  return cat8(lo, hh);
}

__global__ void __launch_bounds__(256)
fattn_fwd(const float* __restrict__ Q, const float* __restrict__ K,
          const float* __restrict__ V, float* __restrict__ O) {
  const int tid   = threadIdx.x;
  const int lane  = tid & 31;
  const int wid   = tid >> 5;
  const int lan16 = lane & 15;
  const int hi    = lane >> 4;

  const int bh = blockIdx.x >> 4;
  const int qt = ((blockIdx.x & 15) << 3) | wid;
  const int b  = bh / NH;
  const int h  = bh % NH;

  const float* kbase = K + (size_t)b * SQ * ROWSTRIDE + h * HD;
  const float* vbase = V + (size_t)b * SQ * ROWSTRIDE + h * HD;
  const float* qbase = Q + (size_t)b * SQ * ROWSTRIDE + h * HD;
  float*       obase = O + ((size_t)b * NH + h) * SQ * HD;

  __shared__ __align__(16) _Float16 Ks[2][TS][KPAD];
  __shared__ __align__(16) _Float16 Vs[2][HD][VPAD];

  const int q0 = qt * 16;
  const float qscale = 0.125f * 1.44269504f;
  const float* qrow = qbase + (size_t)(q0 + lan16) * ROWSTRIDE;
  const v16h bQ0 = load_run16(qrow + 0  + 8 * hi, qscale);
  const v16h bQ1 = load_run16(qrow + 32 + 8 * hi, qscale);

  const int srow = tid >> 3;
  const int sd   = (tid & 7) << 3;
  const float* kptr = kbase + (size_t)srow * ROWSTRIDE + sd;
  const float* vptr = vbase + (size_t)srow * ROWSTRIDE + sd;

  {
    float4 k0 = *(const float4*)(kptr);
    float4 k1 = *(const float4*)(kptr + 4);
    float4 w0 = *(const float4*)(vptr);
    float4 w1 = *(const float4*)(vptr + 4);
    *(h8*)&Ks[0][srow][sd] = cvt8(k0, k1);
    h8 vv = cvt8(w0, w1);
#pragma unroll
    for (int j = 0; j < 8; ++j) Vs[0][sd + j][srow] = vv[j];
  }

  v8f acc[4] = {{0}, {0}, {0}, {0}};
  float m_run = -1.0e30f;
  float l_run = 0.0f;
  const float inv_keep = 1.0f / 0.9f;
  const unsigned seed_base = ((((unsigned)bh << 11) | (unsigned)(q0 + lan16)) << 11);

  int p = 0;
#pragma unroll 1
  for (int i = 0; i < NIT; ++i) {
    const int t0 = i * TS;
    __syncthreads();
    const bool hasnext = (i + 1 < NIT);
    float4 k0, k1, w0, w1;
    if (hasnext) {
      const float* kn = kptr + (size_t)(t0 + TS) * ROWSTRIDE;
      const float* vn = vptr + (size_t)(t0 + TS) * ROWSTRIDE;
      k0 = *(const float4*)(kn);
      k1 = *(const float4*)(kn + 4);
      w0 = *(const float4*)(vn);
      w1 = *(const float4*)(vn + 4);
    }

    const _Float16* kr0 = &Ks[p][lan16][0];
    const _Float16* kr1 = &Ks[p][16 + lan16][0];
    v8f s0 = {}, s1 = {};
    s0 = __builtin_amdgcn_wmma_f32_16x16x32_f16(false, ldsA(kr0, 0, hi),  false, bQ0, (short)0, s0, false, false);
    s0 = __builtin_amdgcn_wmma_f32_16x16x32_f16(false, ldsA(kr0, 32, hi), false, bQ1, (short)0, s0, false, false);
    s1 = __builtin_amdgcn_wmma_f32_16x16x32_f16(false, ldsA(kr1, 0, hi),  false, bQ0, (short)0, s1, false, false);
    s1 = __builtin_amdgcn_wmma_f32_16x16x32_f16(false, ldsA(kr1, 32, hi), false, bQ1, (short)0, s1, false, false);

    float mt = s0[0];
#pragma unroll
    for (int r = 1; r < 8; ++r) mt = fmaxf(mt, s0[r]);
#pragma unroll
    for (int r = 0; r < 8; ++r) mt = fmaxf(mt, s1[r]);
    mt = fmaxf(mt, __shfl_xor(mt, 16, 32));
    const float m_new = fmaxf(m_run, mt);
    const float corr  = __builtin_amdgcn_exp2f(m_run - m_new);

    const unsigned h0 = hash32(seed_base | (unsigned)(t0 + 8 * hi));
    const unsigned h1 = hash32(seed_base | (unsigned)(t0 + 8 * hi + 4));
    const unsigned h2 = hash32(seed_base | (unsigned)(t0 + 16 + 8 * hi));
    const unsigned h3 = hash32(seed_base | (unsigned)(t0 + 20 + 8 * hi));

    float lt = 0.0f;
    v16h aP;
#pragma unroll
    for (int r = 0; r < 8; ++r) {
      const float pr = __builtin_amdgcn_exp2f(s0[r] - m_new);
      lt += pr;
      const unsigned hh = (r < 4) ? h0 : h1;
      const float mk = (((hh >> (8 * (r & 3))) & 255u) < 230u) ? inv_keep : 0.0f;
      aP[r] = (_Float16)(pr * mk);
    }
#pragma unroll
    for (int r = 0; r < 8; ++r) {
      const float pr = __builtin_amdgcn_exp2f(s1[r] - m_new);
      lt += pr;
      const unsigned hh = (r < 4) ? h2 : h3;
      const float mk = (((hh >> (8 * (r & 3))) & 255u) < 230u) ? inv_keep : 0.0f;
      aP[8 + r] = (_Float16)(pr * mk);
    }
    lt += __shfl_xor(lt, 16, 32);
    l_run = l_run * corr + lt;
    m_run = m_new;

#pragma unroll
    for (int r = 0; r < 8; ++r) {
      const float c = __shfl(corr, r + 8 * hi, 32);
      acc[0][r] *= c; acc[1][r] *= c; acc[2][r] *= c; acc[3][r] *= c;
    }

#pragma unroll
    for (int dn = 0; dn < 4; ++dn) {
      const _Float16* vrow = &Vs[p][dn * 16 + lan16][8 * hi];
      const v16h bV = cat8(*(const h8*)(vrow), *(const h8*)(vrow + 16));
      acc[dn] = __builtin_amdgcn_wmma_f32_16x16x32_f16(false, aP, false, bV, (short)0,
                                                       acc[dn], false, false);
    }

    if (hasnext) {
      *(h8*)&Ks[p ^ 1][srow][sd] = cvt8(k0, k1);
      h8 vv = cvt8(w0, w1);
#pragma unroll
      for (int j = 0; j < 8; ++j) Vs[p ^ 1][sd + j][srow] = vv[j];
    }
    p ^= 1;
  }

#pragma unroll
  for (int r = 0; r < 8; ++r) {
    const float linv = 1.0f / __shfl(l_run, r + 8 * hi, 32);
    float* orow = obase + (size_t)(q0 + r + 8 * hi) * HD + lan16;
    orow[0]  = acc[0][r] * linv;
    orow[16] = acc[1][r] * linv;
    orow[32] = acc[2][r] * linv;
    orow[48] = acc[3][r] * linv;
  }
}

extern "C" void kernel_launch(void* const* d_in, const int* in_sizes, int n_in,
                              void* d_out, int out_size, void* d_ws, size_t ws_size,
                              hipStream_t stream) {
  const float* q = (const float*)d_in[0];
  const float* k = (const float*)d_in[1];
  const float* v = (const float*)d_in[2];
  float* o = (float*)d_out;

  const size_t elems = (size_t)NBH * SQ * HD;          // per tensor, f16
  const size_t need  = 2 * elems * sizeof(_Float16);   // Kh + Vt = 12.58 MB

  if (ws_size >= need) {
    _Float16* Kh = (_Float16*)d_ws;
    _Float16* Vt = Kh + elems;
    preconv<<<dim3(SQ / 64, NBH), 256, 0, stream>>>(k, v, Kh, Vt);
    fattn_f16<<<(NBH * 128) / 8, 256, 0, stream>>>(q, Kh, Vt, o);
  } else {
    fattn_fwd<<<NBH * 16, 256, 0, stream>>>(q, k, v, o);
  }
}
